// CustomNARX_82497731822067
// MI455X (gfx1250) — compile-verified, matrix operand order checked by reading, not run on hardware
//
#include <hip/hip_runtime.h>

// CDNA5 / gfx1250 NARX scan kernel.
// Each wave32 owns a 16-row batch tile and runs the full 128-step recurrence
// with all state in LDS. Series history kept in a 64-slot ring (slot j = t mod
// 64) so WMMA A-fragments are step-invariant, 16B-aligned ds_load_b128 pairs;
// the K-rotation is folded into precomputed per-phase weight fragments in
// global scratch (L2-resident, ~2MB). All GEMMs lower to
// v_wmma_f32_16x16x32_f16. Sigmoid: v_fma + v_exp_f32 + v_rcp_f32; h staged
// to LDS with packed v_cvt_pk_f16_f32 + ds_store_b32 under a K-permuted
// layout (W3 fragments permuted to match).

typedef __attribute__((ext_vector_type(16))) _Float16 v16h;
typedef __attribute__((ext_vector_type(8)))  _Float16 v8h;
typedef __attribute__((ext_vector_type(2)))  __fp16   v2hp;   // cvt_pkrtz result type
typedef __attribute__((ext_vector_type(8)))  float    v8f;

#define RSTR      72                    // ring row stride (f16): 64 + pad (bank-conflict free, 16B mult)
#define HSTR      168                   // h-buffer row stride (f16): 160 + pad (16B mult)
#define RING_SER  (16 * RSTR)           // 1152 f16 per series per wave
#define WAVE_LDS  (5 * RING_SER + 16 * HSTR)   // 5760 + 2688 = 8448 f16 per wave
#define NEG_L2E   (-1.44269504088896340736f)

// f16 A/B fragment element -> K mapping (ISA 7.12.2, 16-bit A 16x32 layout)
__device__ __forceinline__ int kval(int lh, int e) {
  int j = e >> 1;                                   // VGPR index 0..7
  return ((j >> 2) << 4) + ((j & 3) << 1) + (e & 1) + lh * 8;
}

// ---------------- prep: build rotated, fragment-ordered f16 weights ----------------
// W2R: [64 phase][5 series][2 kfrag][2 nfrag][32 lane][16 e]   (655360 f16)
// W1R: [64 phase][5 series][2 kfrag][32 lane][16 e]            (327680 f16)
// W3F: [5 kfrag][32 lane][16 e]                                (  2560 f16)
__global__ void narx_prep(const float* __restrict__ W1, const float* __restrict__ W2,
                          const float* __restrict__ W3,
                          _Float16* __restrict__ W2R, _Float16* __restrict__ W1R,
                          _Float16* __restrict__ W3F) {
  int tid = blockIdx.x * blockDim.x + threadIdx.x;
  if (tid < 40960) {                                // W2R fragments
    int L = tid & 31; int rest = tid >> 5;
    int q  = rest & 1; rest >>= 1;
    int kf = rest & 1; rest >>= 1;
    int i  = rest % 5; int p = rest / 5;
    int lm = L & 15, lh = L >> 4, n = q * 16 + lm;
    v16h v;
#pragma unroll
    for (int e = 0; e < 16; ++e) {
      int jj = kf * 32 + kval(lh, e);               // ring slot = K index
      int t  = (jj - p) & 63;                       // rotate by phase
      v[e] = (_Float16)W2[i * 2048 + n * 64 + t];   // W2: (5,32,64)
    }
    *(v16h*)(W2R + (size_t)tid * 16) = v;
  } else if (tid < 40960 + 20480) {                 // W1R fragments (out1, N padded to 16)
    int t2 = tid - 40960;
    int L = t2 & 31; int rest = t2 >> 5;
    int kf = rest & 1; rest >>= 1;
    int i  = rest % 5; int p = rest / 5;
    int lm = L & 15, lh = L >> 4;
    v16h v;
#pragma unroll
    for (int e = 0; e < 16; ++e) {
      int jj = kf * 32 + kval(lh, e);
      int t  = (jj - p) & 63;
      v[e] = (lm < 2) ? (_Float16)W1[lm * 320 + t * 5 + i] : (_Float16)0.0f; // W1: (2,320)
    }
    *(v16h*)(W1R + (size_t)t2 * 16) = v;
  } else if (tid < 40960 + 20480 + 160) {           // W3F fragments (out2, static)
    int t3 = tid - 40960 - 20480;
    int L = t3 & 31; int kf5 = t3 >> 5;
    int lm = L & 15, lh = L >> 4;
    v16h v;
#pragma unroll
    for (int e = 0; e < 16; ++e) {
      // K-permuted hbuf layout: k' = i*32 + col*2 + q  <->  k = i*32 + q*16 + col
      int kp   = kf5 * 32 + kval(lh, e);            // permuted K index (matches hbuf)
      int r5   = kp & 31;
      int korg = (kp & ~31) + (r5 & 1) * 16 + (r5 >> 1);
      v[e] = (lm < 2) ? (_Float16)W3[lm * 160 + korg] : (_Float16)0.0f;      // W3: (2,160)
    }
    *(v16h*)(W3F + (size_t)t3 * 16) = v;
  }
}

// ---------------- main: one wave = 16 batch rows, full 128-step scan ----------------
__global__ void __launch_bounds__(64)
narx_main(const float* __restrict__ target, const float* __restrict__ exog,
          const float* __restrict__ b1, const float* __restrict__ b2,
          const float* __restrict__ b3,
          const _Float16* __restrict__ W2R, const _Float16* __restrict__ W1R,
          const _Float16* __restrict__ W3F, float* __restrict__ out) {
  __shared__ __align__(16) _Float16 smem[2 * WAVE_LDS];
  const int w    = threadIdx.x >> 5;
  const int lane = threadIdx.x & 31;
  const int lm   = lane & 15, lh = lane >> 4;
  const int b0   = blockIdx.x * 32 + w * 16;

  _Float16* ring = smem + w * WAVE_LDS;             // 5 series x 16 rows x RSTR
  _Float16* hbuf = ring + 5 * RING_SER;             // 16 rows x HSTR

  // ---- initial ring fill: slot j = t for t in [0,64) ----
  for (int idx = lane; idx < 2048; idx += 32) {     // targets: series 0,1
    int i = idx >> 10, m = (idx >> 6) & 15, j = idx & 63;
    ring[i * RING_SER + m * RSTR + j] = (_Float16)target[(size_t)(b0 + m) * 128 + j * 2 + i];
  }
  for (int idx = lane; idx < 3072; idx += 32) {     // exog: series 2,3,4
    int i = idx >> 10, m = (idx >> 6) & 15, j = idx & 63;
    ring[(2 + i) * RING_SER + m * RSTR + j] = (_Float16)exog[(size_t)(b0 + m) * 576 + j * 3 + i];
  }

  // ---- per-lane constants (bias pre-scaled by -log2(e) for fused fma) ----
  float b2s[5][2];
#pragma unroll
  for (int i = 0; i < 5; ++i)
#pragma unroll
    for (int q = 0; q < 2; ++q) b2s[i][q] = b2[i * 32 + q * 16 + lm] * NEG_L2E;
  const float bias13 = (lm < 2) ? (b1[lm] + b3[lm]) : 0.0f;
  v16h w3f[5];
#pragma unroll
  for (int kf = 0; kf < 5; ++kf) w3f[kf] = *(const v16h*)(W3F + kf * 512 + lane * 16);

  // window[:,-1,:] carried in registers (f32): lanes lm<2 hold rows r=0..7
  float lastv[8];
  if (lm < 2) {
#pragma unroll
    for (int r = 0; r < 8; ++r)
      lastv[r] = target[(size_t)(b0 + r + lh * 8) * 128 + 63 * 2 + lm];
  }

  // ---- sequential scan ----
#pragma unroll 1
  for (int s = 0; s < 128; ++s) {
    const int p = s & 63;
    const _Float16* w2p = W2R + (size_t)p * 10240;  // 5*2*2*512
    const _Float16* w1p = W1R + (size_t)p * 5120;   // 5*2*512

    v8f c1 = {};                                    // out1 accumulator (N: 2 valid of 16)
#pragma unroll
    for (int i = 0; i < 5; ++i) {
      v8f ch0 = {}, ch1 = {};                       // h for this series, nfrag 0/1
#pragma unroll
      for (int kf = 0; kf < 2; ++kf) {
        const _Float16* ar = ring + i * RING_SER + lm * RSTR + kf * 32 + lh * 8;
        v8h alo = *(const v8h*)(ar);                // ds_load_b128, 16B aligned
        v8h ahi = *(const v8h*)(ar + 16);
        v16h a;
#pragma unroll
        for (int e = 0; e < 8; ++e) { a[e] = alo[e]; a[8 + e] = ahi[e]; }
        v16h bh0 = *(const v16h*)(w2p + ((i * 2 + kf) * 2 + 0) * 512 + lane * 16);
        v16h bh1 = *(const v16h*)(w2p + ((i * 2 + kf) * 2 + 1) * 512 + lane * 16);
        v16h bw1 = *(const v16h*)(w1p + (i * 2 + kf) * 512 + lane * 16);
        ch0 = __builtin_amdgcn_wmma_f32_16x16x32_f16(false, a, false, bh0, (short)0, ch0, false, false);
        ch1 = __builtin_amdgcn_wmma_f32_16x16x32_f16(false, a, false, bh1, (short)0, ch1, false, false);
        c1  = __builtin_amdgcn_wmma_f32_16x16x32_f16(false, a, false, bw1, (short)0, c1,  false, false);
      }
      // sigmoid(h + b2): t = fma(c, -log2e, b2*-log2e); rcp(1+exp2(t)).
      // Pack (q=0,q=1) pair -> one ds_store_b32 at permuted column lm*2.
#pragma unroll
      for (int r = 0; r < 8; ++r) {
        float t0 = __builtin_fmaf(ch0[r], NEG_L2E, b2s[i][0]);
        float t1 = __builtin_fmaf(ch1[r], NEG_L2E, b2s[i][1]);
        float s0 = __builtin_amdgcn_rcpf(1.0f + __builtin_amdgcn_exp2f(t0));
        float s1 = __builtin_amdgcn_rcpf(1.0f + __builtin_amdgcn_exp2f(t1));
        v2hp pk = __builtin_amdgcn_cvt_pkrtz(s0, s1);
        *(v2hp*)(hbuf + (r + lh * 8) * HSTR + i * 32 + lm * 2) = pk;
      }
    }

    // out2 = sigmoid(h) @ W3^T  (K=160, permuted layout; static B fragments)
    v8f c2 = {};
#pragma unroll
    for (int kf = 0; kf < 5; ++kf) {
      const _Float16* hr = hbuf + lm * HSTR + kf * 32 + lh * 8;
      v8h alo = *(const v8h*)(hr);
      v8h ahi = *(const v8h*)(hr + 16);
      v16h a;
#pragma unroll
      for (int e = 0; e < 8; ++e) { a[e] = alo[e]; a[8 + e] = ahi[e]; }
      c2 = __builtin_amdgcn_wmma_f32_16x16x32_f16(false, a, false, w3f[kf], (short)0, c2, false, false);
    }

    // pred = out1 + out2 + window[:,-1,:]; emit output; append into ring
    const int slot_new = s & 63;
    if (lm < 2) {
#pragma unroll
      for (int r = 0; r < 8; ++r) {
        int m = r + lh * 8;
        float pv = c1[r] + c2[r] + bias13 + lastv[r];
        lastv[r] = pv;                              // window[:,-1,:] for step s+1
        out[(size_t)(b0 + m) * 256 + s * 2 + lm] = pv;
        ring[lm * RING_SER + m * RSTR + slot_new] = (_Float16)pv;
      }
    }
    // slide exog: bring in t = s + 64 (<= 191) into the freed slot
#pragma unroll
    for (int rr = 0; rr < 2; ++rr) {
      int idx = lane + rr * 32;
      if (idx < 48) {
        int m = idx & 15, i = idx >> 4;
        float v = exog[(size_t)(b0 + m) * 576 + (s + 64) * 3 + i];
        ring[(2 + i) * RING_SER + m * RSTR + slot_new] = (_Float16)v;
      }
    }
  }
}

extern "C" void kernel_launch(void* const* d_in, const int* in_sizes, int n_in,
                              void* d_out, int out_size, void* d_ws, size_t ws_size,
                              hipStream_t stream) {
  const float* target = (const float*)d_in[0];
  const float* exog   = (const float*)d_in[1];
  const float* W1     = (const float*)d_in[2];
  const float* b1     = (const float*)d_in[3];
  const float* W2     = (const float*)d_in[4];
  const float* b2     = (const float*)d_in[5];
  const float* W3     = (const float*)d_in[6];
  const float* b3     = (const float*)d_in[7];
  float* out = (float*)d_out;

  _Float16* W2R = (_Float16*)d_ws;                  // 655360 f16
  _Float16* W1R = W2R + 655360;                     // 327680 f16
  _Float16* W3F = W1R + 327680;                     //   2560 f16  (total ~1.93 MB)

  narx_prep<<<(61600 + 255) / 256, 256, 0, stream>>>(W1, W2, W3, W2R, W1R, W3F);
  // 8192 rows / (2 waves * 16 rows) = 256 workgroups of 64 threads (2 wave32)
  narx_main<<<256, 64, 0, stream>>>(target, exog, b1, b2, b3, W2R, W1R, W3F, out);
}
